// VanillaRNN_56727928045615
// MI455X (gfx1250) — matrix operand order repaired and checked by reading, not compiled, not run on hardware
//
#include <hip/hip_runtime.h>
#include <hip/hip_bf16.h>

// Problem constants (VanillaRNN: x (64,512,1024), W (1024,1024))
constexpr int NSEQ   = 64;
constexpr int TSTEPS = 512;
constexpr int DH     = 1024;                // D == H == 1024
constexpr int MTOT   = NSEQ * TSTEPS;       // 32768 rows for phase-1 GEMM

typedef __attribute__((ext_vector_type(16))) __bf16         v16bf;
typedef __attribute__((ext_vector_type(8)))  float          v8f;
typedef __attribute__((ext_vector_type(4)))  unsigned int   uv4;
typedef __attribute__((ext_vector_type(4)))  unsigned short us4;

// WMMA fragment viewed as two 16-byte loads (matches CDNA5 bf16 A/B VGPR layout)
union Frag { v16bf v; uv4 q[2]; };

__device__ __forceinline__ unsigned short f2bf(float f) {
    unsigned u = __builtin_bit_cast(unsigned, f);
    u += 0x7FFFu + ((u >> 16) & 1u);        // round-to-nearest-even
    return (unsigned short)(u >> 16);
}

// ---------------------------------------------------------------------------
// Kernel 0a: x fp32 -> bf16 (row-major, A-operand friendly: K contiguous)
// ---------------------------------------------------------------------------
__global__ void cvt_x(const float* __restrict__ x, unsigned short* __restrict__ xb) {
    size_t i = ((size_t)blockIdx.x * blockDim.x + threadIdx.x) * 4;
    const float4 f = *(const float4*)(x + i);
    us4 o;
    o[0] = f2bf(f.x); o[1] = f2bf(f.y); o[2] = f2bf(f.z); o[3] = f2bf(f.w);
    *(us4*)(xb + i) = o;
}

// ---------------------------------------------------------------------------
// Kernel 0b: W (k,n) fp32 -> W^T (n,k) bf16, for both Wxh and Whh.
// B-operand wants K contiguous per output column n.
// ---------------------------------------------------------------------------
__global__ void cvt_wT(const float* __restrict__ wxh, const float* __restrict__ whh,
                       unsigned short* __restrict__ wxhT, unsigned short* __restrict__ whhT) {
    int i = blockIdx.x * 256 + threadIdx.x;          // 2 * 1024 * 1024 elements
    int which = i >> 20;
    int e = i & (DH * DH - 1);
    int k = e >> 10, n = e & (DH - 1);
    const float* src        = which ? whh  : wxh;
    unsigned short* dst     = which ? whhT : wxhT;
    dst[(size_t)n * DH + k] = f2bf(src[(size_t)k * DH + n]);
}

// ---------------------------------------------------------------------------
// Phase 1: xh = x @ Wxh + (bias_xh + bias_hh), written fp32 into d_out.
// Block = 256 threads = 8 wave32, block tile 128x128, wave tile 32x64.
// Fragments read directly from global as b128 (x_bf16 lives in L2: 64 MB).
// Software-pipelined: K-step k0+32 fragments are in flight while k0 computes.
// ---------------------------------------------------------------------------
__global__ __launch_bounds__(256) void gemm_xh(
        const unsigned short* __restrict__ xb,    // 32768 x 1024 bf16
        const unsigned short* __restrict__ wxhT,  // [n][k] 1024 x 1024 bf16
        const float* __restrict__ bias_xh,
        const float* __restrict__ bias_hh,
        float* __restrict__ out) {
    const int lane = threadIdx.x & 31, w = threadIdx.x >> 5;
    const int l16 = lane & 15, g = lane >> 4;
    const int wm = w & 3, wn = w >> 2;               // 4 waves in M, 2 in N
    const int bM = blockIdx.x * 128, bN = blockIdx.y * 128;
    const int colB0 = bN + wn * 64;

    // Loop-invariant, lane-resolved base pointers (include g-dependent chunk offs)
    const unsigned short* ap[2];
    ap[0] = xb + (size_t)(bM + wm * 32 + l16) * DH + g * 8;
    ap[1] = xb + (size_t)(bM + wm * 32 + 16 + l16) * DH + g * 8;
    const unsigned short* bp[4];
#pragma unroll
    for (int nt = 0; nt < 4; ++nt)
        bp[nt] = wxhT + (size_t)(colB0 + nt * 16 + l16) * DH + g * 8;

    v8f acc[2][4] = {};
    Frag A[2], B[4];

    // prologue: k0 = 0 fragments
#pragma unroll
    for (int mt = 0; mt < 2; ++mt) {
        A[mt].q[0] = *(const uv4*)(ap[mt]);
        A[mt].q[1] = *(const uv4*)(ap[mt] + 16);
    }
#pragma unroll
    for (int nt = 0; nt < 4; ++nt) {
        B[nt].q[0] = *(const uv4*)(bp[nt]);
        B[nt].q[1] = *(const uv4*)(bp[nt] + 16);
    }

    for (int k0 = 32; k0 < DH; k0 += 32) {
        Frag nA[2], nB[4];
        // prefetch next K-step while current WMMAs execute
#pragma unroll
        for (int mt = 0; mt < 2; ++mt) {
            nA[mt].q[0] = *(const uv4*)(ap[mt] + k0);
            nA[mt].q[1] = *(const uv4*)(ap[mt] + k0 + 16);
        }
#pragma unroll
        for (int nt = 0; nt < 4; ++nt) {
            nB[nt].q[0] = *(const uv4*)(bp[nt] + k0);
            nB[nt].q[1] = *(const uv4*)(bp[nt] + k0 + 16);
        }
#pragma unroll
        for (int nt = 0; nt < 4; ++nt)
#pragma unroll
            for (int mt = 0; mt < 2; ++mt)
                acc[mt][nt] = __builtin_amdgcn_wmma_f32_16x16x32_bf16(
                    false, A[mt].v, false, B[nt].v, (short)0, acc[mt][nt], false, false);
#pragma unroll
        for (int mt = 0; mt < 2; ++mt) A[mt] = nA[mt];
#pragma unroll
        for (int nt = 0; nt < 4; ++nt) B[nt] = nB[nt];
    }
    // epilogue K-step
#pragma unroll
    for (int nt = 0; nt < 4; ++nt)
#pragma unroll
        for (int mt = 0; mt < 2; ++mt)
            acc[mt][nt] = __builtin_amdgcn_wmma_f32_16x16x32_bf16(
                false, A[mt].v, false, B[nt].v, (short)0, acc[mt][nt], false, false);

#pragma unroll
    for (int nt = 0; nt < 4; ++nt) {
        const int col = colB0 + nt * 16 + l16;
        const float bs = bias_xh[col] + bias_hh[col];
#pragma unroll
        for (int mt = 0; mt < 2; ++mt) {
            const int row = bM + wm * 32 + mt * 16 + g * 8;  // D layout: M = g*8 + v
#pragma unroll
            for (int v = 0; v < 8; ++v)
                out[(size_t)(row + v) * DH + col] = acc[mt][nt][v] + bs;
        }
    }
}

// ---------------------------------------------------------------------------
// Phase 2: persistent recurrence. 32 blocks x 256 threads; block owns 32
// hidden columns; WhhT slice (32 x 1024 bf16, padded stride 1032) lives in
// LDS for all 512 steps. h double-buffered in bf16 in workspace; one
// monotonic-counter grid barrier per timestep. Inner K-loop software-pipelined.
// ---------------------------------------------------------------------------
__global__ __launch_bounds__(256) void rnn_scan(
        const unsigned short* __restrict__ whhT,  // [n][k] 1024 x 1024 bf16
        unsigned short* __restrict__ hbuf,        // 2 x 64 x 1024 bf16 (zeroed)
        float* __restrict__ xh_y,                 // (64,512,1024): xh in, y out
        unsigned* __restrict__ counter) {
    extern __shared__ unsigned short lds[];       // 32 x 1032 bf16
    constexpr int LSTR = DH + 8;                  // pad: 2064 B rows, 16B aligned
    const int lane = threadIdx.x & 31, w = threadIdx.x >> 5;
    const int l16 = lane & 15, g = lane >> 4;
    const int mt = w & 3, nt = w >> 2;            // 4 M-tiles x 2 N-tiles
    const int c0 = blockIdx.x * 32;

    // Stage this block's 32 columns of Whh^T into LDS (b128 copies).
    for (int idx = threadIdx.x; idx < 32 * 128; idx += 256) {
        int r = idx >> 7, c = (idx & 127) * 8;
        *(uv4*)(lds + r * LSTR + c) = *(const uv4*)(whhT + (size_t)(c0 + r) * DH + c);
    }
    __syncthreads();

    const int m0   = mt * 16;
    const int col  = c0 + nt * 16 + l16;
    const int mrow = m0 + g * 8;                             // D layout row base
    const size_t aoff = (size_t)(m0 + l16) * DH + g * 8;     // lane A base offset
    const unsigned short* bbase = lds + (size_t)(nt * 16 + l16) * LSTR + g * 8;

    for (int t = 0; t < TSTEPS; ++t) {
        const unsigned short* hin  = hbuf + (size_t)(t & 1) * NSEQ * DH;
        unsigned short*       hout = hbuf + (size_t)((t + 1) & 1) * NSEQ * DH;

        // acc starts as this tile of xh[:, t, :]
        v8f acc;
#pragma unroll
        for (int v = 0; v < 8; ++v)
            acc[v] = xh_y[((size_t)(mrow + v) * TSTEPS + t) * DH + col];

        const unsigned short* apb = hin + aoff;
        Frag A, B;
        A.q[0] = *(const uv4*)(apb);
        A.q[1] = *(const uv4*)(apb + 16);
        B.q[0] = *(const uv4*)(bbase);
        B.q[1] = *(const uv4*)(bbase + 16);
        for (int k0 = 32; k0 < DH; k0 += 32) {
            Frag nA, nB;
            nA.q[0] = *(const uv4*)(apb + k0);
            nA.q[1] = *(const uv4*)(apb + k0 + 16);
            nB.q[0] = *(const uv4*)(bbase + k0);
            nB.q[1] = *(const uv4*)(bbase + k0 + 16);
            acc = __builtin_amdgcn_wmma_f32_16x16x32_bf16(
                false, A.v, false, B.v, (short)0, acc, false, false);
            A = nA; B = nB;
        }
        acc = __builtin_amdgcn_wmma_f32_16x16x32_bf16(
            false, A.v, false, B.v, (short)0, acc, false, false);

#pragma unroll
        for (int v = 0; v < 8; ++v) {
            float y = tanhf(acc[v]);
            xh_y[((size_t)(mrow + v) * TSTEPS + t) * DH + col] = y;  // in-place y
            hout[(size_t)(mrow + v) * DH + col] = f2bf(y);           // next-step h
        }

        // ---- grid barrier (monotonic counter; double buffer -> 1 sync/step)
        __threadfence();
        __syncthreads();
        if (threadIdx.x == 0) {
            __hip_atomic_fetch_add(counter, 1u, __ATOMIC_RELEASE,
                                   __HIP_MEMORY_SCOPE_AGENT);
            const unsigned target = (unsigned)gridDim.x * (unsigned)(t + 1);
            while (__hip_atomic_load(counter, __ATOMIC_ACQUIRE,
                                     __HIP_MEMORY_SCOPE_AGENT) < target)
                __builtin_amdgcn_s_sleep(2);
        }
        __syncthreads();
        __threadfence();
    }
}

// ---------------------------------------------------------------------------
extern "C" void kernel_launch(void* const* d_in, const int* in_sizes, int n_in,
                              void* d_out, int out_size, void* d_ws, size_t ws_size,
                              hipStream_t stream) {
    const float* x   = (const float*)d_in[0];
    const float* wxh = (const float*)d_in[1];
    const float* whh = (const float*)d_in[2];
    const float* bxh = (const float*)d_in[3];
    const float* bhh = (const float*)d_in[4];
    float* out = (float*)d_out;

    char* ws = (char*)d_ws;
    // workspace layout (bytes)
    unsigned short* xb   = (unsigned short*)(ws);                 // 64 MiB
    unsigned short* wxhT = (unsigned short*)(ws + 67108864);      // 2 MiB
    unsigned short* whhT = (unsigned short*)(ws + 69206016);      // 2 MiB
    unsigned short* hbuf = (unsigned short*)(ws + 71303168);      // 256 KiB
    unsigned*      count = (unsigned*)     (ws + 71565312);       // 4 B

    // zero h double-buffer + sync counter (graph-capture-safe async memset)
    hipMemsetAsync(hbuf, 0, (size_t)2 * NSEQ * DH * 2 + 64, stream);

    cvt_x <<<(MTOT * DH) / (256 * 4), 256, 0, stream>>>(x, xb);
    cvt_wT<<<(2 * DH * DH) / 256,     256, 0, stream>>>(wxh, whh, wxhT, whhT);

    dim3 g1(MTOT / 128, DH / 128);                                // 256 x 8
    gemm_xh<<<g1, 256, 0, stream>>>(xb, wxhT, bxh, bhh, out);

    const size_t ldsBytes = (size_t)32 * (DH + 8) * 2;            // 66048 B
    rnn_scan<<<32, 256, ldsBytes, stream>>>(whhT, hbuf, out, count);
}
